// MultHeadAttention_34557306864208
// MI455X (gfx1250) — compile-verified, matrix operand order checked by reading, not run on hardware
//
#include <hip/hip_runtime.h>
#include <hip/hip_bf16.h>
#include <cstdint>

#define SEQ    4096
#define EMBED  1024
#define NHEAD  16
#define HD     64
#define F3     3072   // 3*EMBED

typedef __attribute__((ext_vector_type(16))) __bf16 v16bf;
typedef __attribute__((ext_vector_type(8)))  float  v8f;
typedef uint32_t u32x4 __attribute__((ext_vector_type(4)));
typedef uint32_t u32x8 __attribute__((ext_vector_type(8)));

union AFrag {            // 16 bf16 = one WMMA A/B operand (8 VGPRs)
    v16bf v;
    uint4 q[2];
};
union FAcc {             // 16x16 f32 C/D tile (8 VGPRs)
    v8f   v;
    float f[8];
};

__device__ __forceinline__ uint16_t bfbits(float f) {
    union { float f; uint32_t u; } x; x.f = f;
    uint32_t r = x.u + 0x7FFFu + ((x.u >> 16) & 1u);   // round-to-nearest-even
    return (uint16_t)(r >> 16);
}
__device__ __forceinline__ uint32_t f2bf2(float a, float b) {
    return (uint32_t)bfbits(a) | ((uint32_t)bfbits(b) << 16);
}

// ---- CDNA5 async global -> LDS (ASYNCcnt path), per-lane 16B ---------------
__device__ __forceinline__ uint32_t lds_off(const void* p) {
    return (uint32_t)(size_t)p;            // low 32 bits of shared-aperture addr = LDS offset
}
__device__ __forceinline__ void async_load_b128(uint32_t lds_addr, const void* gptr) {
    asm volatile("global_load_async_to_lds_b128 %0, %1, off"
                 :: "v"(lds_addr), "v"((uint64_t)(size_t)gptr)
                 : "memory");
}
__device__ __forceinline__ void wait_async0() {
    asm volatile("s_wait_asynccnt 0x0" ::: "memory");
}

// ---------------------------------------------------------------------------
// Kernel 0: fp32 -> bf16 streaming conversion (x, w_qkv, w_out)
// ---------------------------------------------------------------------------
__global__ __launch_bounds__(256)
void cvt_f32_bf16(const float* __restrict__ src, uint16_t* __restrict__ dst, int n) {
    int i = (blockIdx.x * 256 + threadIdx.x) * 4;
    if (i + 3 < n) {
        float4 f = *(const float4*)(src + i);
        uint2 p;
        p.x = f2bf2(f.x, f.y);
        p.y = f2bf2(f.z, f.w);
        *(uint2*)(dst + i) = p;
    }
}

// ---------------------------------------------------------------------------
// Kernel 1: QKV projection  qkv[s,f] = sum_e x[s,e] * w_qkv[f,e] + b_qkv[f]
// Workgroup tile 64(M) x 256(N); 8 waves (2x4), each 32x64 (2x4 WMMA tiles).
// Double-buffered async global->LDS staging: DMA tile i+1 overlaps WMMA on i.
// ---------------------------------------------------------------------------
__global__ __launch_bounds__(256)
void qkv_gemm(const uint16_t* __restrict__ xb,   // [SEQ][EMBED] bf16
              const uint16_t* __restrict__ wb,   // [F3][EMBED]  bf16
              const float*    __restrict__ bias, // [F3]
              uint16_t* __restrict__ Qw,         // [NHEAD][SEQ][HD]
              uint16_t* __restrict__ Kw,         // [NHEAD][SEQ][HD]
              uint16_t* __restrict__ Vt)         // [NHEAD][HD][SEQ]
{
    __shared__ __attribute__((aligned(16))) uint16_t As[2][64][32];    //  8 KB
    __shared__ __attribute__((aligned(16))) uint16_t Bs[2][256][32];   // 32 KB

    const int t   = threadIdx.x;
    const int w   = t >> 5, l = t & 31;
    const int wm  = w >> 2, wn = w & 3;            // 2 x 4 wave grid
    const int m0  = blockIdx.x * 64;
    const int n0  = blockIdx.y * 256;
    const int col = l & 15;
    const int rhi = (l >> 4) << 3;                 // 0 or 8 (C-row offset)
    const int klo = (l >> 4) << 3;                 // K lo/hi half per lane group

    const int aRow = t >> 2, aCol = (t & 3) << 3;

    auto issue = [&](int k0, int buf) {
        async_load_b128(lds_off(&As[buf][aRow][aCol]),
                        &xb[(m0 + aRow) * EMBED + k0 + aCol]);
#pragma unroll
        for (int i = 0; i < 4; ++i) {              // B tile 256x32: 4 x 16B per thread
            int c = t + i * 256;
            int row = c >> 2, cg = (c & 3) << 3;
            async_load_b128(lds_off(&Bs[buf][row][cg]),
                            &wb[(n0 + row) * EMBED + k0 + cg]);
        }
    };

    FAcc acc[2][4] = {};

    issue(0, 0);
    wait_async0();
    __syncthreads();

    for (int k0 = 0; k0 < EMBED; k0 += 32) {
        const int cur = (k0 >> 5) & 1;
        if (k0 + 32 < EMBED) issue(k0 + 32, cur ^ 1);

        AFrag a[2], b[4];
#pragma unroll
        for (int mt = 0; mt < 2; ++mt) {
            int r = wm * 32 + mt * 16 + col;
            a[mt].q[0] = *(const uint4*)&As[cur][r][klo];
            a[mt].q[1] = *(const uint4*)&As[cur][r][klo + 16];
        }
#pragma unroll
        for (int nt = 0; nt < 4; ++nt) {
            int r = wn * 64 + nt * 16 + col;
            b[nt].q[0] = *(const uint4*)&Bs[cur][r][klo];
            b[nt].q[1] = *(const uint4*)&Bs[cur][r][klo + 16];
        }
#pragma unroll
        for (int mt = 0; mt < 2; ++mt)
#pragma unroll
            for (int nt = 0; nt < 4; ++nt)
                acc[mt][nt].v = __builtin_amdgcn_wmma_f32_16x16x32_bf16(
                    false, a[mt].v, false, b[nt].v, (short)0, acc[mt][nt].v, false, false);

        wait_async0();
        __syncthreads();
    }

    // Epilogue: f = h*192 + chunk*64 + d.  f0 is wave-uniform, so head/chunk
    // math is uniform; 16-wide tiles never straddle q/k/v or head boundaries.
#pragma unroll
    for (int mt = 0; mt < 2; ++mt)
#pragma unroll
        for (int nt = 0; nt < 4; ++nt) {
            int f0    = n0 + wn * 64 + nt * 16;
            int h     = f0 / 192;
            int rem   = f0 % 192;
            int chunk = rem / 64;
            int d     = (rem % 64) + col;
            float bv  = bias[f0 + col];
            int sbase = m0 + wm * 32 + mt * 16 + rhi;
            if (chunk == 0) {
#pragma unroll
                for (int g = 0; g < 8; ++g)
                    Qw[((h * SEQ) + sbase + g) * HD + d] = bfbits(acc[mt][nt].f[g] + bv);
            } else if (chunk == 1) {
#pragma unroll
                for (int g = 0; g < 8; ++g)
                    Kw[((h * SEQ) + sbase + g) * HD + d] = bfbits(acc[mt][nt].f[g] + bv);
            } else {   // V stored transposed: 8 consecutive s -> one 16B store
                uint4 p;
                p.x = f2bf2(acc[mt][nt].f[0] + bv, acc[mt][nt].f[1] + bv);
                p.y = f2bf2(acc[mt][nt].f[2] + bv, acc[mt][nt].f[3] + bv);
                p.z = f2bf2(acc[mt][nt].f[4] + bv, acc[mt][nt].f[5] + bv);
                p.w = f2bf2(acc[mt][nt].f[6] + bv, acc[mt][nt].f[7] + bv);
                *(uint4*)&Vt[((h * HD) + d) * SEQ + sbase] = p;
            }
        }
}

// ---------------------------------------------------------------------------
// Kernel 2: flash attention. Workgroup = 1 head x 128 queries (8 waves x 16).
// Double-buffered async K/V staging; WMMA scores; online softmax;
// P transposed through LDS; WMMA P.V.
// ---------------------------------------------------------------------------
__global__ __launch_bounds__(256)
void flash_attn(const uint16_t* __restrict__ Qw,
                const uint16_t* __restrict__ Kw,
                const uint16_t* __restrict__ Vt,
                uint16_t* __restrict__ Ob)        // [SEQ][EMBED] bf16
{
    __shared__ __attribute__((aligned(16))) uint16_t Ks[2][32][64];   // 8 KB
    __shared__ __attribute__((aligned(16))) uint16_t Vs[2][64][32];   // 8 KB
    __shared__ __attribute__((aligned(16))) uint16_t Ps[8][16][32];   // 8 KB

    const int h   = blockIdx.x;
    const int qb  = blockIdx.y;
    const int t   = threadIdx.x, w = t >> 5, l = t & 31;
    const int col = l & 15;
    const int rhi = (l >> 4) << 3;
    const int klo = (l >> 4) << 3;
    const int q0  = qb * 128 + w * 16;
    const float scale = 0.125f;                    // 1/sqrt(64)

    // Persistent Q fragments (16 x 64 -> two K=32 A-frags)
    AFrag aq[2];
    {
        const uint16_t* qrow = Qw + (size_t)(h * SEQ + q0 + col) * HD;
#pragma unroll
        for (int ks = 0; ks < 2; ++ks) {
            aq[ks].q[0] = *(const uint4*)(qrow + ks * 32 + klo);
            aq[ks].q[1] = *(const uint4*)(qrow + ks * 32 + klo + 16);
        }
    }

    FAcc o[4] = {};
    float mr[8], lr[8];
#pragma unroll
    for (int g = 0; g < 8; ++g) { mr[g] = -3.0e38f; lr[g] = 0.0f; }

    const int kRow = t >> 3, kCol = (t & 7) << 3;  // K-tile coop coords
    const int vRow = t >> 2, vCol = (t & 3) << 3;  // V-tile coop coords

    auto issue = [&](int kb, int buf) {
        async_load_b128(lds_off(&Ks[buf][kRow][kCol]),
                        &Kw[((size_t)h * SEQ + kb * 32 + kRow) * HD + kCol]);
        async_load_b128(lds_off(&Vs[buf][vRow][vCol]),
                        &Vt[((size_t)h * HD + vRow) * SEQ + kb * 32 + vCol]);
    };

    issue(0, 0);
    wait_async0();
    __syncthreads();

    const int NB = SEQ / 32;
    for (int kb = 0; kb < NB; ++kb) {
        const int cur = kb & 1;
        if (kb + 1 < NB) {
            issue(kb + 1, cur ^ 1);
            __builtin_prefetch(&Kw[((size_t)h * SEQ + (kb + 2) * 32 + kRow) * HD + kCol], 0, 0);
        }

        // Scores: S[16 x 32] = Q(16x64) . K_blk^T(64x32)
        FAcc s[2] = {};
#pragma unroll
        for (int kk = 0; kk < 2; ++kk)
#pragma unroll
            for (int ks = 0; ks < 2; ++ks) {
                AFrag bk;
                int r = kk * 16 + col;
                bk.q[0] = *(const uint4*)&Ks[cur][r][ks * 32 + klo];
                bk.q[1] = *(const uint4*)&Ks[cur][r][ks * 32 + klo + 16];
                s[kk].v = __builtin_amdgcn_wmma_f32_16x16x32_bf16(
                    false, aq[ks].v, false, bk.v, (short)0, s[kk].v, false, false);
            }

        // Online softmax, row-wise (rows live in 16-lane halves)
#pragma unroll
        for (int g = 0; g < 8; ++g) {
            float a = s[0].f[g] * scale;
            float b = s[1].f[g] * scale;
            float tm = fmaxf(a, b);
            tm = fmaxf(tm, __shfl_xor(tm, 1));
            tm = fmaxf(tm, __shfl_xor(tm, 2));
            tm = fmaxf(tm, __shfl_xor(tm, 4));
            tm = fmaxf(tm, __shfl_xor(tm, 8));
            float mn = fmaxf(mr[g], tm);
            float ef = __expf(mr[g] - mn);
            float pa = __expf(a - mn);
            float pb = __expf(b - mn);
            float ps = pa + pb;
            ps += __shfl_xor(ps, 1);
            ps += __shfl_xor(ps, 2);
            ps += __shfl_xor(ps, 4);
            ps += __shfl_xor(ps, 8);
            lr[g] = lr[g] * ef + ps;
            mr[g] = mn;
#pragma unroll
            for (int n = 0; n < 4; ++n) o[n].f[g] *= ef;
            Ps[w][g + rhi][col]      = bfbits(pa);   // C-layout -> LDS
            Ps[w][g + rhi][col + 16] = bfbits(pb);
        }

        // Re-read P in A-layout (per-wave transpose through LDS)
        AFrag ap;
        ap.q[0] = *(const uint4*)&Ps[w][col][klo];
        ap.q[1] = *(const uint4*)&Ps[w][col][klo + 16];

        // O += P(16x32) . V_blk(32x64)  (Vs is [d][key], so column d contiguous)
#pragma unroll
        for (int n = 0; n < 4; ++n) {
            AFrag bv;
            int r = n * 16 + col;
            bv.q[0] = *(const uint4*)&Vs[cur][r][klo];
            bv.q[1] = *(const uint4*)&Vs[cur][r][klo + 16];
            o[n].v = __builtin_amdgcn_wmma_f32_16x16x32_bf16(
                false, ap.v, false, bv.v, (short)0, o[n].v, false, false);
        }

        wait_async0();
        __syncthreads();
    }

    // Normalize and store O as bf16 [s][h*64+d] for the output projection
#pragma unroll
    for (int n = 0; n < 4; ++n) {
        int d = n * 16 + col;
#pragma unroll
        for (int g = 0; g < 8; ++g) {
            float v = o[n].f[g] / lr[g];
            Ob[(size_t)(q0 + g + rhi) * EMBED + h * HD + d] = bfbits(v);
        }
    }
}

// ---------------------------------------------------------------------------
// Kernel 3: output projection. Workgroup tile 64x256, wave tile 32x64.
// A tile via async->LDS; B tile via Tensor Data Mover (double-buffered).
// ---------------------------------------------------------------------------
__global__ __launch_bounds__(256)
void out_gemm(const uint16_t* __restrict__ Ob,   // [SEQ][EMBED] bf16
              const uint16_t* __restrict__ wb,   // [EMBED][EMBED] bf16
              const float*    __restrict__ bias, // [EMBED]
              float* __restrict__ out)           // [SEQ][EMBED] f32
{
    __shared__ __attribute__((aligned(16))) uint16_t As[2][64][32];    //  8 KB
    __shared__ __attribute__((aligned(16))) uint16_t Bs[2][256][32];   // 32 KB

    const int t   = threadIdx.x;
    const int w   = t >> 5, l = t & 31;
    const int wm  = w >> 2, wn = w & 3;
    const int m0  = blockIdx.x * 64;
    const int n0  = blockIdx.y * 256;
    const int col = l & 15;
    const int rhi = (l >> 4) << 3;
    const int klo = (l >> 4) << 3;

    const int aRow = t >> 2, aCol = (t & 3) << 3;

    auto issue = [&](int k0, int buf) {
        // A tile 64x32: async per-lane
        async_load_b128(lds_off(&As[buf][aRow][aCol]),
                        &Ob[(size_t)(m0 + aRow) * EMBED + k0 + aCol]);
        // B tile 256x32 via TDM: wave 0 issues one DMA descriptor
        if (t < 32) {
            uint64_t gaddr = (uint64_t)(size_t)(wb + (size_t)n0 * EMBED + k0);
            u32x4 g0;
            g0[0] = 1u;                                         // count=1, user descriptor
            g0[1] = lds_off(&Bs[buf][0][0]);                    // lds_addr
            g0[2] = (uint32_t)gaddr;                            // global_addr[31:0]
            g0[3] = (uint32_t)((gaddr >> 32) & 0x01FFFFFFu)     // global_addr[56:32]
                    | (2u << 30);                               // type = 2 ("image")
            u32x8 g1;
            g1[0] = 1u << 16;                                   // data_size = 1 -> 2 bytes
            g1[1] = (uint32_t)(EMBED & 0xFFFFu) << 16;          // tensor_dim0[15:0]
            g1[2] = (uint32_t)(EMBED >> 16)                     // tensor_dim0[31:16]
                    | ((uint32_t)(EMBED & 0xFFFFu) << 16);      // tensor_dim1[15:0]
            g1[3] = (uint32_t)(EMBED >> 16) | (32u << 16);      // tensor_dim1 hi | tile_dim0=32
            g1[4] = 256u;                                       // tile_dim1=256, tile_dim2=0
            g1[5] = (uint32_t)EMBED;                            // tensor_dim0_stride lo
            g1[6] = 0u;
            g1[7] = 0u;
            asm volatile("tensor_load_to_lds %0, %1" :: "s"(g0), "s"(g1) : "memory");
        }
    };

    FAcc acc[2][4] = {};

    issue(0, 0);
    if (t < 32) __builtin_amdgcn_s_wait_tensorcnt(0);
    wait_async0();
    __syncthreads();

    for (int k0 = 0; k0 < EMBED; k0 += 32) {
        const int cur = (k0 >> 5) & 1;
        if (k0 + 32 < EMBED) issue(k0 + 32, cur ^ 1);

        AFrag a[2], b[4];
#pragma unroll
        for (int mt = 0; mt < 2; ++mt) {
            int r = wm * 32 + mt * 16 + col;
            a[mt].q[0] = *(const uint4*)&As[cur][r][klo];
            a[mt].q[1] = *(const uint4*)&As[cur][r][klo + 16];
        }
#pragma unroll
        for (int nt = 0; nt < 4; ++nt) {
            int r = wn * 64 + nt * 16 + col;
            b[nt].q[0] = *(const uint4*)&Bs[cur][r][klo];
            b[nt].q[1] = *(const uint4*)&Bs[cur][r][klo + 16];
        }
#pragma unroll
        for (int mt = 0; mt < 2; ++mt)
#pragma unroll
            for (int nt = 0; nt < 4; ++nt)
                acc[mt][nt].v = __builtin_amdgcn_wmma_f32_16x16x32_bf16(
                    false, a[mt].v, false, b[nt].v, (short)0, acc[mt][nt].v, false, false);

        if (t < 32) __builtin_amdgcn_s_wait_tensorcnt(0);
        wait_async0();
        __syncthreads();
    }

#pragma unroll
    for (int mt = 0; mt < 2; ++mt)
#pragma unroll
        for (int nt = 0; nt < 4; ++nt) {
            int n    = n0 + wn * 64 + nt * 16 + col;
            float bv = bias[n];
            int  sb  = m0 + wm * 32 + mt * 16 + rhi;
#pragma unroll
            for (int g = 0; g < 8; ++g)
                out[(size_t)(sb + g) * EMBED + n] = acc[mt][nt].f[g] + bv;
        }
}

// ---------------------------------------------------------------------------
extern "C" void kernel_launch(void* const* d_in, const int* in_sizes, int n_in,
                              void* d_out, int out_size, void* d_ws, size_t ws_size,
                              hipStream_t stream) {
    const float* x     = (const float*)d_in[0];   // [1,4096,1024]
    const float* w_qkv = (const float*)d_in[1];   // [3072,1024]
    const float* b_qkv = (const float*)d_in[2];   // [3072]
    const float* w_out = (const float*)d_in[3];   // [1024,1024]
    const float* b_out = (const float*)d_in[4];   // [1024]
    float* out = (float*)d_out;

    char* ws = (char*)d_ws;
    size_t off = 0;
    auto carve = [&](size_t elems) { uint16_t* p = (uint16_t*)(ws + off); off += elems * 2; return p; };
    uint16_t* xb    = carve((size_t)SEQ * EMBED);    // 8 MB
    uint16_t* wqkvb = carve((size_t)F3 * EMBED);     // 6 MB
    uint16_t* woutb = carve((size_t)EMBED * EMBED);  // 2 MB
    uint16_t* Qw    = carve((size_t)SEQ * EMBED);    // 8 MB  [h][s][64]
    uint16_t* Kw    = carve((size_t)SEQ * EMBED);    // 8 MB  [h][s][64]
    uint16_t* Vt    = carve((size_t)SEQ * EMBED);    // 8 MB  [h][64][s]
    uint16_t* Ob    = carve((size_t)SEQ * EMBED);    // 8 MB  [s][e]
    (void)ws_size;

    cvt_f32_bf16<<<(SEQ * EMBED) / 1024, 256, 0, stream>>>(x, xb, SEQ * EMBED);
    cvt_f32_bf16<<<(F3 * EMBED) / 1024, 256, 0, stream>>>(w_qkv, wqkvb, F3 * EMBED);
    cvt_f32_bf16<<<(EMBED * EMBED) / 1024, 256, 0, stream>>>(w_out, woutb, EMBED * EMBED);

    qkv_gemm<<<dim3(SEQ / 64, F3 / 256), 256, 0, stream>>>(xb, wqkvb, b_qkv, Qw, Kw, Vt);

    flash_attn<<<dim3(NHEAD, SEQ / 128), 256, 0, stream>>>(Qw, Kw, Vt, Ob);

    out_gemm<<<dim3(SEQ / 64, EMBED / 128 / 2), 256, 0, stream>>>(Ob, woutb, b_out, out);
}
